// BStarMemory_37950331028263
// MI455X (gfx1250) — matrix operand order repaired and checked by reading, not compiled, not run on hardware
//
#include <hip/hip_runtime.h>
#include <math.h>

#define DIM     1024
#define NKEYS   65536
#define TOKENS  4096
#define TOPK    32
#define HID     512
#define NCHUNKS 16                 // key chunks
#define KCHUNK  4096               // keys per workgroup (kernel 3)
#define MTILE   64                 // tokens per workgroup (kernel 3)
#define MSUB    4                  // m-subtiles of 16
#define SLAB    256                // keys per GEMM slab (8 waves * 32 keys)
#define NSLABS  (KCHUNK / SLAB)
#define CAND_PER_TOKEN (NCHUNKS * TOPK)   // 512
#define QCAP    64                 // LDS candidate queue capacity per token
#define NEG_INF (-3.4e38f)

typedef __attribute__((ext_vector_type(16))) __bf16 v16bf;
typedef __attribute__((ext_vector_type(8)))  float  v8f;

union Frag { v16bf v; unsigned u[8]; };

// ---- fp32 -> (bf16_hi, bf16_lo) truncation split (error lives in lo) ----
__device__ __forceinline__ void split1(float f, unsigned short& h, unsigned short& l) {
    unsigned u = __float_as_uint(f);
    h = (unsigned short)(u >> 16);
    float lo = f - __uint_as_float(u & 0xFFFF0000u);
    l = (unsigned short)(__float_as_uint(lo) >> 16);
}

__device__ __forceinline__ void split2(float f0, float f1, unsigned& hp, unsigned& lp) {
    unsigned u0 = __float_as_uint(f0), u1 = __float_as_uint(f1);
    hp = (u0 >> 16) | (u1 & 0xFFFF0000u);
    float l0 = f0 - __uint_as_float(u0 & 0xFFFF0000u);
    float l1 = f1 - __uint_as_float(u1 & 0xFFFF0000u);
    unsigned lu0 = __float_as_uint(l0), lu1 = __float_as_uint(l1);
    lp = (lu0 >> 16) | (lu1 & 0xFFFF0000u);
}

// ---- A fragment (16x32 bf16) from an LDS row-major bf16 plane, row stride DIM ----
__device__ __forceinline__ void load_a_lds(const unsigned short* plane, int lane, int kb, Frag& f) {
    int r = lane & 15, h = (lane >> 4) & 1;
    const unsigned short* p = plane + r * DIM + kb + h * 8;
    uint4 a = *(const uint4*)(p);
    uint4 b = *(const uint4*)(p + 16);
    f.u[0] = a.x; f.u[1] = a.y; f.u[2] = a.z; f.u[3] = a.w;
    f.u[4] = b.x; f.u[5] = b.y; f.u[6] = b.z; f.u[7] = b.w;
}

// ---- B fragment (32x16 bf16) from a pre-converted global bf16 plane [N][DIM] ----
__device__ __forceinline__ void load_b_bf(const unsigned short* __restrict__ plane,
                                          int nbase, int lane, int kb, Frag& f) {
    size_t n = (size_t)(nbase + (lane & 15));
    int h = (lane >> 4) & 1;
    const unsigned short* p = plane + n * DIM + kb + h * 8;
    uint4 a = *(const uint4*)(p);
    uint4 b = *(const uint4*)(p + 16);
    f.u[0] = a.x; f.u[1] = a.y; f.u[2] = a.z; f.u[3] = a.w;
    f.u[4] = b.x; f.u[5] = b.y; f.u[6] = b.z; f.u[7] = b.w;
}

// ---- 3-WMMA split-precision accumulate: acc += (ah+al) x (bh+bl), drop al*bl ----
__device__ __forceinline__ v8f wmma3(const Frag& ah, const Frag& al,
                                     const Frag& bh, const Frag& bl, v8f c) {
    c = __builtin_amdgcn_wmma_f32_16x16x32_bf16(false, al.v, false, bh.v, (short)0, c, false, false);
    c = __builtin_amdgcn_wmma_f32_16x16x32_bf16(false, ah.v, false, bl.v, (short)0, c, false, false);
    c = __builtin_amdgcn_wmma_f32_16x16x32_bf16(false, ah.v, false, bh.v, (short)0, c, false, false);
    return c;
}

// ---- async 16B global -> LDS copy (ASYNCcnt-tracked; ISA 15.18.3 op 98) ----
// dst must be an LDS pointer; low 32 bits of the generic pointer are the DS offset.
__device__ __forceinline__ void async_copy16(void* lds_dst, const void* gsrc) {
    unsigned dsaddr = (unsigned)(unsigned long long)(uintptr_t)lds_dst;
    unsigned long long gaddr = (unsigned long long)(uintptr_t)gsrc;
    asm volatile("global_load_async_to_lds_b128 %0, %1, off"
                 :: "v"(dsaddr), "v"(gaddr) : "memory");
}
__device__ __forceinline__ void async_wait0() {
    asm volatile("s_wait_asynccnt 0" ::: "memory");
}

// =================== kernel 0: fp32 matrix -> split bf16 planes (one-time) ===================
__global__ __launch_bounds__(256) void k_cvt(const float* __restrict__ src,
                                             unsigned short* __restrict__ hi,
                                             unsigned short* __restrict__ lo,
                                             int count8) {
    int i = blockIdx.x * 256 + threadIdx.x;
    if (i < count8) {
        const float4* p = (const float4*)(src + (size_t)i * 8);
        float4 a = p[0], b = p[1];
        uint4 h4, l4;
        split2(a.x, a.y, h4.x, l4.x);
        split2(a.z, a.w, h4.y, l4.y);
        split2(b.x, b.y, h4.z, l4.z);
        split2(b.z, b.w, h4.w, l4.w);
        ((uint4*)hi)[i] = h4;
        ((uint4*)lo)[i] = l4;
    }
}

// =================== kernel 1: per-key exploration bonus ===================
__global__ __launch_bounds__(256) void k_bonus(const float* __restrict__ acc,
                                               const float* __restrict__ succ,
                                               float* __restrict__ bonus) {
    int n = blockIdx.x * 256 + threadIdx.x;
    if (n < NKEYS) {
        float a = acc[n];
        float sr = succ[n] / (a + 1e-10f);
        float ex = 1.0f / (a + 1.0f);
        bonus[n] = 0.1f * (0.9f * sr + 0.1f * ex);
    }
}

// =================== kernel 2: query projection -> split bf16 query planes ===================
__global__ __launch_bounds__(256) void k_qproj(const float* __restrict__ x,
                                               const unsigned short* __restrict__ wqhi,
                                               const unsigned short* __restrict__ wqlo,
                                               const float* __restrict__ bq,
                                               unsigned short* __restrict__ qhiG,
                                               unsigned short* __restrict__ qloG) {
    extern __shared__ unsigned char smem[];
    unsigned short* xhi = (unsigned short*)smem;       // [16][DIM]
    unsigned short* xlo = xhi + 16 * DIM;              // [16][DIM]
    const int tid = threadIdx.x, lane = tid & 31, wave = tid >> 5;
    const size_t rowBase = (size_t)blockIdx.x * 16;

#pragma unroll 4
    for (int i = 0; i < 64; ++i) {                     // stage 16x1024 split into LDS
        int e = tid + 256 * i, r = e >> 10, c = e & (DIM - 1);
        split1(x[(rowBase + r) * DIM + c], xhi[r * DIM + c], xlo[r * DIM + c]);
    }
    __syncthreads();

    v8f zero = {0.f, 0.f, 0.f, 0.f, 0.f, 0.f, 0.f, 0.f};
    v8f acc[8];
#pragma unroll
    for (int i = 0; i < 8; ++i) acc[i] = zero;

    for (int ks = 0; ks < 32; ++ks) {
        int kb = ks * 32;
        Frag ah, al;
        load_a_lds(xhi, lane, kb, ah);
        load_a_lds(xlo, lane, kb, al);
#pragma unroll
        for (int nt = 0; nt < 8; ++nt) {
            Frag bh, bl;
            load_b_bf(wqhi, wave * 128 + nt * 16, lane, kb, bh);
            load_b_bf(wqlo, wave * 128 + nt * 16, lane, kb, bl);
            acc[nt] = wmma3(ah, al, bh, bl, acc[nt]);
        }
    }
    const int hi = lane >> 4, nc = lane & 15;
#pragma unroll
    for (int nt = 0; nt < 8; ++nt) {
        int e = wave * 128 + nt * 16 + nc;
        float b = bq[e];
#pragma unroll
        for (int r = 0; r < 8; ++r) {
            size_t off = (rowBase + r + 8 * hi) * DIM + e;
            split1(acc[nt][r] + b, qhiG[off], qloG[off]);
        }
    }
}

// =================== kernel 3: scores GEMM + chunk-local top-32 ===================
__global__ __launch_bounds__(256) void k_scores(const unsigned short* __restrict__ qhiG,
                                                const unsigned short* __restrict__ qloG,
                                                const unsigned short* __restrict__ khiG,
                                                const unsigned short* __restrict__ kloG,
                                                const float* __restrict__ bonus,
                                                float* __restrict__ candS,
                                                int* __restrict__ candI) {
    extern __shared__ unsigned char smem[];
    unsigned short* qhi = (unsigned short*)smem;            // [64][DIM] 128 KB
    unsigned short* qlo = qhi + MTILE * DIM;                // [64][DIM] 128 KB
    float* qS   = (float*)(qlo + MTILE * DIM);              // [64][QCAP]
    int*   qK   = (int*)  (qS + MTILE * QCAP);              // [64][QCAP]
    float* lS   = (float*)(qK + MTILE * QCAP);              // [64][TOPK]
    int*   lI   = (int*)  (lS + MTILE * TOPK);              // [64][TOPK]
    int*   lCnt = (int*)  (lI + MTILE * TOPK);              // [64]
    float* lMin = (float*)(lCnt + MTILE);                   // [64]
    int*   lMnP = (int*)  (lMin + MTILE);                   // [64]
    int*   qN   = (int*)  (lMnP + MTILE);                   // [64]
    int*   ovf  = (int*)  (qN + MTILE);                     // [1]

    const int tid = threadIdx.x, lane = tid & 31, wave = tid >> 5;
    const size_t rowBase = (size_t)blockIdx.x * MTILE;
    const int chunk = blockIdx.y;

    {   // stage split query tile via async global->LDS copies (ASYNCcnt path)
        const uint4* srcH = (const uint4*)(qhiG + rowBase * DIM);
        const uint4* srcL = (const uint4*)(qloG + rowBase * DIM);
        uint4* dstH = (uint4*)qhi;
        uint4* dstL = (uint4*)qlo;
        for (int i = tid; i < MTILE * DIM / 8; i += 256) {
            async_copy16(dstH + i, srcH + i);
            async_copy16(dstL + i, srcL + i);
        }
        async_wait0();
    }
    if (tid < MTILE) { lCnt[tid] = 0; lMin[tid] = NEG_INF; lMnP[tid] = 0; qN[tid] = 0; }
    if (tid == 0) *ovf = 0;
    __syncthreads();

    const int hi = lane >> 4, nc = lane & 15;
    v8f zero = {0.f, 0.f, 0.f, 0.f, 0.f, 0.f, 0.f, 0.f};

    for (int s = 0; s < NSLABS; ++s) {
        const int keyBase = chunk * KCHUNK + s * SLAB + wave * 32;
        if (s + 1 < NSLABS) {   // hint next slab of key planes into cache
            __builtin_prefetch(khiG + (size_t)(keyBase + SLAB) * DIM + lane * 64, 0, 1);
            __builtin_prefetch(kloG + (size_t)(keyBase + SLAB) * DIM + lane * 64, 0, 1);
        }
        v8f acc[MSUB][2];
#pragma unroll
        for (int m = 0; m < MSUB; ++m) { acc[m][0] = zero; acc[m][1] = zero; }

#pragma unroll 2
        for (int ks = 0; ks < 32; ++ks) {
            int kb = ks * 32;
            Frag b0h, b0l, b1h, b1l;
            load_b_bf(khiG, keyBase,      lane, kb, b0h);
            load_b_bf(kloG, keyBase,      lane, kb, b0l);
            load_b_bf(khiG, keyBase + 16, lane, kb, b1h);
            load_b_bf(kloG, keyBase + 16, lane, kb, b1l);
#pragma unroll
            for (int m = 0; m < MSUB; ++m) {
                Frag ah, al;
                load_a_lds(qhi + m * 16 * DIM, lane, kb, ah);
                load_a_lds(qlo + m * 16 * DIM, lane, kb, al);
                acc[m][0] = wmma3(ah, al, b0h, b0l, acc[m][0]);
                acc[m][1] = wmma3(ah, al, b1h, b1l, acc[m][1]);
            }
        }
        const int n0 = keyBase + nc, n1 = keyBase + 16 + nc;
        const float b0 = bonus[n0], b1 = bonus[n1];
#pragma unroll
        for (int m = 0; m < MSUB; ++m)
#pragma unroll
            for (int r = 0; r < 8; ++r) {
                acc[m][0][r] = acc[m][0][r] * 0.03125f + b0;   // 1/sqrt(1024) + bonus
                acc[m][1][r] = acc[m][1][r] * 0.03125f + b1;
            }

        unsigned long long pend = ~0ull;   // 64 scores per lane: (msub, ntile, r)
        while (true) {
            // -------- push phase: threshold filter into per-token LDS queues --------
#pragma unroll
            for (int i = 0; i < 64; ++i) {
                if ((pend >> i) & 1ull) {
                    int ms = i >> 4, nt = (i >> 3) & 1, r = i & 7;
                    int mTok = ms * 16 + r + 8 * hi;
                    float sv = acc[ms][nt][r];
                    bool qual = (lCnt[mTok] < TOPK) || (sv > lMin[mTok]);
                    if (!qual) { pend &= ~(1ull << i); }
                    else {
                        int pos = atomicAdd(&qN[mTok], 1);
                        if (pos < QCAP) {
                            qS[mTok * QCAP + pos] = sv;
                            qK[mTok * QCAP + pos] = nt ? n1 : n0;
                            pend &= ~(1ull << i);
                        } else { *ovf = 1; }
                    }
                }
            }
            __syncthreads();
            // -------- drain phase: one thread per token merges queue into top-32 ----
            if (tid < MTILE) {
                int m = tid;
                int take = qN[m]; if (take > QCAP) take = QCAP;
                int c = lCnt[m]; float cm = lMin[m]; int mp = lMnP[m];
                for (int j = 0; j < take; ++j) {
                    float sv = qS[m * QCAP + j]; int kv = qK[m * QCAP + j];
                    if (c < TOPK) {
                        lS[m * TOPK + c] = sv; lI[m * TOPK + c] = kv; ++c;
                        if (c == TOPK) {
                            cm = lS[m * TOPK]; mp = 0;
                            for (int a = 1; a < TOPK; ++a) {
                                float v = lS[m * TOPK + a];
                                if (v < cm) { cm = v; mp = a; }
                            }
                        }
                    } else if (sv > cm) {
                        lS[m * TOPK + mp] = sv; lI[m * TOPK + mp] = kv;
                        cm = lS[m * TOPK]; mp = 0;
                        for (int a = 1; a < TOPK; ++a) {
                            float v = lS[m * TOPK + a];
                            if (v < cm) { cm = v; mp = a; }
                        }
                    }
                }
                lCnt[m] = c; lMin[m] = cm; lMnP[m] = mp; qN[m] = 0;
            }
            __syncthreads();
            int done = (*ovf == 0);
            __syncthreads();
            if (done) break;
            if (tid == 0) *ovf = 0;
            __syncthreads();
        }
    }
    // -------- emit 32 candidates per token for this chunk --------
    for (int e = tid; e < MTILE * TOPK; e += 256) {
        int m = e >> 5, j = e & 31;
        size_t off = ((rowBase + m) * NCHUNKS + chunk) * TOPK + j;
        if (j < lCnt[m]) { candS[off] = lS[m * TOPK + j]; candI[off] = lI[m * TOPK + j]; }
        else             { candS[off] = NEG_INF;          candI[off] = 0; }
    }
}

// =================== kernel 4: merge candidates -> sorted top-32 + softmax ===================
__global__ __launch_bounds__(256) void k_merge(const float* __restrict__ candS,
                                               const int* __restrict__ candI,
                                               float* __restrict__ weights,
                                               int* __restrict__ idxOut) {
    int t = blockIdx.x * 256 + threadIdx.x;
    float bs[TOPK]; int bi[TOPK];
    int cnt = 0; float cm = NEG_INF; int mp = 0;
    const float* cs = candS + (size_t)t * CAND_PER_TOKEN;
    const int*   ci = candI + (size_t)t * CAND_PER_TOKEN;
    for (int j = 0; j < CAND_PER_TOKEN; ++j) {
        float s = cs[j];
        if (cnt < TOPK) {
            bs[cnt] = s; bi[cnt] = ci[j]; ++cnt;
            if (cnt == TOPK) {
                cm = bs[0]; mp = 0;
                for (int a = 1; a < TOPK; ++a) if (bs[a] < cm) { cm = bs[a]; mp = a; }
            }
        } else if (s > cm) {
            bs[mp] = s; bi[mp] = ci[j];
            cm = bs[0]; mp = 0;
            for (int a = 1; a < TOPK; ++a) if (bs[a] < cm) { cm = bs[a]; mp = a; }
        }
    }
    for (int a = 0; a < TOPK - 1; ++a) {           // sort descending, ties -> lower index
        int best = a;
        for (int b = a + 1; b < TOPK; ++b)
            if (bs[b] > bs[best] || (bs[b] == bs[best] && bi[b] < bi[best])) best = b;
        float ts = bs[a]; bs[a] = bs[best]; bs[best] = ts;
        int   ti = bi[a]; bi[a] = bi[best]; bi[best] = ti;
    }
    float mx = bs[0], sum = 0.f, w[TOPK];
    for (int a = 0; a < TOPK; ++a) { w[a] = expf(bs[a] - mx); sum += w[a]; }
    float inv = 1.0f / sum;
    for (int a = 0; a < TOPK; ++a) {
        weights[(size_t)t * TOPK + a] = w[a] * inv;
        idxOut[(size_t)t * TOPK + a]  = bi[a];
    }
}

// =================== kernel 5: weighted value gather ===================
__global__ __launch_bounds__(256) void k_gather(const float* __restrict__ values,
                                                const float* __restrict__ weights,
                                                const int* __restrict__ idx,
                                                float* __restrict__ out) {
    __shared__ float  w[TOPK];
    __shared__ int    id[TOPK];
    const int t = blockIdx.x, tid = threadIdx.x;
    if (tid < TOPK) { w[tid] = weights[(size_t)t * TOPK + tid]; id[tid] = idx[(size_t)t * TOPK + tid]; }
    __syncthreads();
    float ax = 0.f, ay = 0.f, az = 0.f, aw = 0.f;
    const int col = tid * 4;
#pragma unroll 8
    for (int k = 0; k < TOPK; ++k) {
        const float4 v = *(const float4*)(values + (size_t)id[k] * DIM + col);
        float wk = w[k];
        ax += wk * v.x; ay += wk * v.y; az += wk * v.z; aw += wk * v.w;
    }
    float4 o; o.x = ax; o.y = ay; o.z = az; o.w = aw;
    *(float4*)(out + (size_t)t * DIM + col) = o;
}

// =================== kernel 6: confidence head ===================
__global__ __launch_bounds__(256) void k_conf(const float* __restrict__ out,
                                              const float* __restrict__ W1c,
                                              const float* __restrict__ b1c,
                                              const float* __restrict__ W2c,
                                              const float* __restrict__ b2c,
                                              float* __restrict__ conf) {
    __shared__ float4 orow[DIM / 4];
    __shared__ float  red[256];
    const int t = blockIdx.x, tid = threadIdx.x;
    orow[tid] = *(const float4*)(out + (size_t)t * DIM + tid * 4);
    __syncthreads();
    float partial = 0.f;
#pragma unroll
    for (int rr = 0; rr < 2; ++rr) {
        int h = tid * 2 + rr;
        const float4* wr = (const float4*)(W1c + (size_t)h * DIM);
        float d = 0.f;
        for (int j = 0; j < DIM / 4; ++j) {
            float4 a = wr[j], b = orow[j];
            d += a.x * b.x + a.y * b.y + a.z * b.z + a.w * b.w;
        }
        float hv = d + b1c[h];
        partial += fmaxf(hv, 0.f) * W2c[h];
    }
    red[tid] = partial;
    __syncthreads();
    for (int st = 128; st > 0; st >>= 1) {
        if (tid < st) red[tid] += red[tid + st];
        __syncthreads();
    }
    if (tid == 0) conf[t] = 1.0f / (1.0f + expf(-(red[0] + b2c[0])));
}

// =================== host launch ===================
extern "C" void kernel_launch(void* const* d_in, const int* in_sizes, int n_in,
                              void* d_out, int out_size, void* d_ws, size_t ws_size,
                              hipStream_t stream) {
    (void)in_sizes; (void)n_in; (void)out_size; (void)ws_size;
    const float* x     = (const float*)d_in[0];
    const float* keys  = (const float*)d_in[1];
    const float* vals  = (const float*)d_in[2];
    const float* Wq    = (const float*)d_in[3];
    const float* bq    = (const float*)d_in[4];
    const float* W1c   = (const float*)d_in[5];
    const float* b1c   = (const float*)d_in[6];
    const float* W2c   = (const float*)d_in[7];
    const float* b2c   = (const float*)d_in[8];
    const float* accC  = (const float*)d_in[9];
    const float* succC = (const float*)d_in[10];

    // workspace layout
    char* ws = (char*)d_ws;
    unsigned short* qhiG = (unsigned short*)ws;                                  // 8 MB
    unsigned short* qloG = qhiG + (size_t)TOKENS * DIM;                          // 8 MB
    unsigned short* khiG = qloG + (size_t)TOKENS * DIM;                          // 128 MB
    unsigned short* kloG = khiG + (size_t)NKEYS * DIM;                           // 128 MB
    unsigned short* wqhi = kloG + (size_t)NKEYS * DIM;                           // 2 MB
    unsigned short* wqlo = wqhi + (size_t)DIM * DIM;                             // 2 MB
    float* bonus   = (float*)(wqlo + (size_t)DIM * DIM);                         // 256 KB
    float* candS   = bonus + NKEYS;                                              // 8 MB
    int*   candI   = (int*)(candS + (size_t)TOKENS * CAND_PER_TOKEN);            // 8 MB
    float* weights = (float*)(candI + (size_t)TOKENS * CAND_PER_TOKEN);          // 512 KB

    float* outF  = (float*)d_out;                                                // [4096,1024]
    float* confF = outF + (size_t)TOKENS * DIM;                                  // [4096]
    int*   idxO  = (int*)(confF + TOKENS);                                       // [4096,32] int32

    const size_t lds_qproj  = 2u * 16 * DIM * sizeof(unsigned short);            // 64 KB
    const size_t lds_scores = 2u * MTILE * DIM * sizeof(unsigned short)          // 256 KB
                            + MTILE * QCAP * (sizeof(float) + sizeof(int))       //  32 KB
                            + MTILE * TOPK * (sizeof(float) + sizeof(int))       //  16 KB
                            + MTILE * 4 * sizeof(int) + sizeof(int);             //  ~1 KB  => ~305 KB

    k_cvt   <<<(NKEYS * DIM / 8 + 255) / 256, 256, 0, stream>>>(keys, khiG, kloG, NKEYS * DIM / 8);
    k_cvt   <<<(DIM * DIM / 8 + 255) / 256, 256, 0, stream>>>(Wq, wqhi, wqlo, DIM * DIM / 8);
    k_bonus <<<NKEYS / 256, 256, 0, stream>>>(accC, succC, bonus);
    k_qproj <<<TOKENS / 16, 256, lds_qproj, stream>>>(x, wqhi, wqlo, bq, qhiG, qloG);
    k_scores<<<dim3(TOKENS / MTILE, NCHUNKS), 256, lds_scores, stream>>>(qhiG, qloG, khiG, kloG,
                                                                        bonus, candS, candI);
    k_merge <<<TOKENS / 256, 256, 0, stream>>>(candS, candI, weights, idxO);
    k_gather<<<TOKENS, 256, 0, stream>>>(vals, weights, idxO, outF);
    k_conf  <<<TOKENS, 256, 0, stream>>>(outF, W1c, b1c, W2c, b2c, confF);
}
